// vlstm_66374424593216
// MI455X (gfx1250) — matrix operand order repaired and checked by reading, not compiled, not run on hardware
//
#include <hip/hip_runtime.h>

// vLSTM for MI455X (gfx1250): per-workgroup recurrence, f16 WMMA GEMMs with
// f32 accumulation, h ping-ponged in LDS, c-state in LDS (wave-exclusive).
//
// B=128, V=32, S=30, P=20, F=8, H=256, gates=4H=1024, OUT=4.
// grid = V * 4 (vessel x 32-batch-row slice), block = 256 threads (8 waves).
// wave w: M-tile m = w&1 (16 batch rows), 4 U-tiles ug+{0,4,8,12}.
// Per (M,U) pair the wave computes all 4 gate tiles (j = g*H + u*16..) so the
// full LSTM cell update happens locally (no cross-wave traffic).
//
// Bias trick: W_ih padded K=8 -> 32 f16; column k=8 carries b_ih+b_hh and
// x[.,8] == 1.0, so the x-WMMA (issued first, C = inline 0) applies the bias.
//
// No local arrays indexed by rolled loop vars -> nothing demoted to scratch.

typedef __attribute__((ext_vector_type(16))) _Float16 v16h;
typedef __attribute__((ext_vector_type(8)))  _Float16 v8h;
typedef __attribute__((ext_vector_type(8)))  float    v8f;

#define B_   128
#define V_   32
#define S_   30
#define P_   20
#define F_   8
#define H_   256
#define G4   1024   // 4*H
#define OUT_ 4
#define KPAD 32     // padded K for the x-GEMM (F=8 -> 32, k=8 = bias column)

union AFrag { v16h v; v8h h[2]; };

__device__ __forceinline__ float sigm(float x) {
    float e = __builtin_amdgcn_exp2f(x * -1.44269504f);
    return __builtin_amdgcn_rcpf(1.0f + e);
}
__device__ __forceinline__ float tanh_fast(float x) {
#if __has_builtin(__builtin_amdgcn_tanhf)
    return __builtin_amdgcn_tanhf(x);          // V_TANH_F32 (gfx1250 trans op)
#else
    float e = __builtin_amdgcn_exp2f(x * -2.88539008f);   // 2^(-2x*log2e)
    return (1.0f - e) * __builtin_amdgcn_rcpf(1.0f + e);
#endif
}

// ---- prep: f32 -> f16 weight conversion (runs every launch, into d_ws) ----
__global__ void cvt_whh_kernel(const float* __restrict__ src,
                               _Float16* __restrict__ dst, int n) {
    int i = blockIdx.x * blockDim.x + threadIdx.x;
    if (i < n) dst[i] = (_Float16)src[i];
}

// W_ih (V,1024,8) f32 -> (V,1024,32) f16; k==8 holds b_ih+b_hh, k>8 zero.
__global__ void cvt_wih_kernel(const float* __restrict__ wih,
                               const float* __restrict__ b_ih,
                               const float* __restrict__ b_hh,
                               _Float16* __restrict__ dst, int n) {
    int i = blockIdx.x * blockDim.x + threadIdx.x;
    if (i >= n) return;
    int k  = i & (KPAD - 1);
    int jv = i >> 5;                 // flat (v*1024 + j)
    float val = 0.0f;
    if (k < F_)       val = wih[jv * F_ + k];
    else if (k == F_) val = b_ih[jv] + b_hh[jv];
    dst[i] = (_Float16)val;
}

// ---- main recurrent kernel ----
__global__ __launch_bounds__(256, 1)
void vlstm_kernel(const float* __restrict__ seq,      // (B,V,S,F)
                  const float* __restrict__ maxval,   // (B,F)
                  const float* __restrict__ minval,   // (B,F)
                  const _Float16* __restrict__ Whh,   // (V,1024,256) f16
                  const _Float16* __restrict__ Wih,   // (V,1024,32)  f16 pad+bias
                  const float* __restrict__ Wlin,     // (F,H)
                  const float* __restrict__ blin,     // (F)
                  float* __restrict__ out)            // (B,V,P,OUT)
{
    __shared__ __align__(16) _Float16 hb[2][32 * H_];   // h ping-pong, 2x16KB
    __shared__ __align__(16) float    cl[32 * H_];      // c state, 32KB
    __shared__ __align__(16) _Float16 xb[32 * KPAD];    // padded x tile, 2KB
    __shared__ __align__(16) float    wl[F_ * H_];      // W_lin copy, 8KB

    const int tid   = threadIdx.x;
    const int lane  = tid & 31;
    const int w     = tid >> 5;          // wave 0..7
    const int v     = blockIdx.x >> 2;   // vessel
    const int mg    = blockIdx.x & 3;    // batch slice
    const int bbase = mg * 32;

    const int m    = w & 1;              // M-tile within WG (0/1)
    const int ug   = w >> 1;             // 0..3 -> U-tiles ug+{0,4,8,12}
    const int half = lane >> 4;          // lane half (K-split selector)
    const int nn   = lane & 15;          // row-in-tile (A) / column (B,C,D)

    // init: x padding (k==8 -> 1.0 bias column), h0 = 0, c0 = 0, W_lin copy
    for (int i = tid; i < 32 * KPAD; i += 256)
        xb[i] = ((i & (KPAD - 1)) == F_) ? (_Float16)1.0f : (_Float16)0.0f;
    for (int i = tid; i < 32 * H_; i += 256) hb[0][i] = (_Float16)0.0f;
    for (int i = tid; i < 32 * H_; i += 256) cl[i] = 0.0f;
    for (int i = tid; i < F_ * H_; i += 256) wl[i] = Wlin[i];

    const _Float16* WhhV = Whh + (size_t)v * G4 * H_;
    const _Float16* WihV = Wih + (size_t)v * G4 * KPAD;

    // x staging / decoder-head assignment: one (row, feature) per thread
    const int   srow  = tid >> 3;        // 0..31
    const int   sf    = tid & 7;         // 0..7
    const int   bglob = bbase + srow;
    const float mx    = maxval[bglob * F_ + sf];
    const float mn    = minval[bglob * F_ + sf];
    const float bl    = blin[sf];
    const float* seqBV = seq + ((size_t)bglob * V_ + v) * S_ * F_;

    int cur = 0;

    for (int step = 0; step < S_ + P_; ++step) {
        // ---- stage x for this step ----
        if (step < S_) {
            xb[srow * KPAD + sf] = (_Float16)seqBV[step * F_ + sf];
        } else if (step == S_) {
            float p0 = seqBV[(S_ - 1) * F_ + sf];
            xb[srow * KPAD + sf] = (_Float16)(p0 * (mx - mn) + mn);
        } // step > S_: xb was written by the decoder epilogue below
        __syncthreads();

        const int nxt = cur ^ 1;

        // A-fragment source row for this wave's M-tile
        // (16x32 A layout: lane half selects K 0..15 / 16..31, matching B)
        const _Float16* hrow = &hb[cur][(m * 16 + nn) * H_];

        AFrag Ax;
        const _Float16* xrow = &xb[(m * 16 + nn) * KPAD];
        Ax.h[0] = *(const v8h*)(xrow + 16 * half);
        Ax.h[1] = *(const v8h*)(xrow + 16 * half + 8);

        #pragma unroll 1            // keep one accumulator set live
        for (int iu = 0; iu < 4; ++iu) {
            const int u = ug + 4 * iu;
            // per-lane base pointers for this U-tile's j = g*H + u*16 + nn
            const _Float16* bh = WhhV + (size_t)(u * 16 + nn) * H_ + 16 * half;
            const _Float16* bx = WihV + (size_t)(u * 16 + nn) * KPAD + 16 * half;

            v8f acc[4];   // only constant-indexed (g unrolled) -> registers
            const v8f zero = {0.f, 0.f, 0.f, 0.f, 0.f, 0.f, 0.f, 0.f};

            // x-GEMM first: applies input term AND bias (k==8 column), C = 0
            #pragma unroll
            for (int g = 0; g < 4; ++g) {
                const _Float16* bp = bx + (size_t)g * H_ * KPAD;
                AFrag Bf;
                Bf.h[0] = *(const v8h*)(bp);
                Bf.h[1] = *(const v8h*)(bp + 8);
                acc[g] = __builtin_amdgcn_wmma_f32_16x16x32_f16(
                    false, Ax.v, false, Bf.v, (short)0, zero, false, false);
            }

            // recurrent GEMM: g += h @ W_hh^T   (K = 256 in 8 chunks of 32)
            // A re-read from LDS each chunk (2x ds_load_b128, no local array)
            // B layout (this code's k order, same as A): lane half selects K 0..15 / 16..31,
            // contiguous in W_hh's k-dim -> two global_load_b128.
            const _Float16* ha = hrow;
            #pragma unroll 1        // limit in-flight B fragments (no spills)
            for (int kc = 0; kc < 8; ++kc) {
                AFrag Af;
                Af.h[0] = *(const v8h*)(ha + 16 * half);
                Af.h[1] = *(const v8h*)(ha + 16 * half + 8);
                ha += 32;
                #pragma unroll
                for (int g = 0; g < 4; ++g) {
                    const _Float16* bp = bh + (size_t)g * H_ * H_;
                    AFrag Bf;
                    Bf.h[0] = *(const v8h*)(bp);
                    Bf.h[1] = *(const v8h*)(bp + 8);
                    acc[g] = __builtin_amdgcn_wmma_f32_16x16x32_f16(
                        false, Af.v, false, Bf.v, (short)0, acc[g],
                        false, false);
                }
                bh += 32;
            }

            // ---- LSTM cell update ----
            // C/D layout: VGPR e, lane -> row = e + 8*half, col = nn.
            // c lives in LDS, exclusively owned by this wave (no barrier).
            float*    cbase = &cl[(m * 16 + 8 * half) * H_ + u * 16 + nn];
            _Float16* hdst  = &hb[nxt][(m * 16 + 8 * half) * H_ + u * 16 + nn];
            #pragma unroll
            for (int e = 0; e < 8; ++e) {
                float ig = sigm(acc[0][e]);
                float fg = sigm(acc[1][e]);
                float gg = tanh_fast(acc[2][e]);
                float og = sigm(acc[3][e]);
                float cn = fg * cbase[e * H_] + ig * gg;
                cbase[e * H_] = cn;
                float hn = og * tanh_fast(cn);
                hdst[e * H_] = (_Float16)hn;
            }
        }
        __syncthreads();
        cur = nxt;

        // ---- decoder epilogue: linear head + next-x staging ----
        if (step >= S_) {
            const int p = step - S_;
            float accd = bl;
            const _Float16* hr = &hb[cur][srow * H_];
            const float*    wr = &wl[sf * H_];
            #pragma unroll
            for (int k = 0; k < H_; k += 8) {
                v8h hv = *(const v8h*)(hr + k);     // ds_load_b128
                #pragma unroll
                for (int e = 0; e < 8; ++e)
                    accd += (float)hv[e] * wr[k + e];
            }
            float r = fminf(fmaxf(accd, 0.0f), 1.0f);
            if (sf < OUT_)
                out[(((size_t)bglob * V_ + v) * P_ + p) * OUT_ + sf] = r;
            xb[srow * KPAD + sf] = (_Float16)(r * (mx - mn) + mn);
            // next iteration's loop-top __syncthreads() covers this write
        }
    }
}

extern "C" void kernel_launch(void* const* d_in, const int* in_sizes, int n_in,
                              void* d_out, int out_size, void* d_ws, size_t ws_size,
                              hipStream_t stream) {
    // setup_inputs order:
    // 0 sequence, 1 dist, 2 rb, 3 cog, 4 seq_mask, 5 maxval, 6 minval,
    // 7 maxVessels, 8 W_ih, 9 W_hh, 10 b_ih, 11 b_hh, 12 W_lin, 13 b_lin
    const float* seq    = (const float*)d_in[0];
    const float* maxval = (const float*)d_in[5];
    const float* minval = (const float*)d_in[6];
    const float* W_ih   = (const float*)d_in[8];
    const float* W_hh   = (const float*)d_in[9];
    const float* b_ih   = (const float*)d_in[10];
    const float* b_hh   = (const float*)d_in[11];
    const float* W_lin  = (const float*)d_in[12];
    const float* b_lin  = (const float*)d_in[13];
    float* out = (float*)d_out;

    // workspace: f16 W_hh (16MB) then padded f16 W_ih+bias (2MB)
    _Float16* Whh_h = (_Float16*)d_ws;
    _Float16* Wih_h = (_Float16*)((char*)d_ws +
                                  (size_t)V_ * G4 * H_ * sizeof(_Float16));

    const int n_whh = V_ * G4 * H_;
    cvt_whh_kernel<<<(n_whh + 255) / 256, 256, 0, stream>>>(W_hh, Whh_h, n_whh);
    const int n_wih = V_ * G4 * KPAD;
    cvt_wih_kernel<<<(n_wih + 255) / 256, 256, 0, stream>>>(W_ih, b_ih, b_hh,
                                                            Wih_h, n_wih);

    vlstm_kernel<<<dim3(V_ * 4), 256, 0, stream>>>(
        seq, maxval, minval, Whh_h, Wih_h, W_lin, b_lin, out);
}
